// FEM_74285754351950
// MI455X (gfx1250) — matrix operand order repaired and checked by reading, not compiled, not run on hardware
//
#include <hip/hip_runtime.h>
#include <hip/hip_bf16.h>
#include <stdint.h>

// ---------------- problem constants ----------------
#define BATCH 8
#define CCH   64
#define NEXP  8
#define HH    128
#define WW    128
// im2col K for 3x3 convs: k = (ky*3+kx)*64 + cin  -> K = 576, 18 k-tiles of 32
#define KT3   18

// use CDNA5 async global->LDS staging in the expert kernel
#define USE_ASYNC_STAGING 1

typedef __bf16           v16bf  __attribute__((ext_vector_type(16)));
typedef float            v8f    __attribute__((ext_vector_type(8)));
typedef float            f32x4  __attribute__((ext_vector_type(4)));
typedef unsigned short   u16x8  __attribute__((ext_vector_type(8)));
typedef unsigned short   u16x16 __attribute__((ext_vector_type(16)));

// ---------------- workspace layout (bytes) ----------------
// h (conv1 output, channel-last bf16): [B][H][W][128]  = 33554432 B
#define OFF_H      0
#define OFF_POOL   33554432                 // [B][64] f32 sums  (2048 B)
#define OFF_EXPW   (OFF_POOL + 2048)        // [B][8]  f32 gate  (256 B)
#define OFF_PC1    (OFF_EXPW + 256)         // conv1 packed frags: 18kt*8nt*1024B = 147456
#define OFF_PE1    (OFF_PC1 + 147456)       // e1 packed: 8e*18kt*4nt*1024B = 589824
#define OFF_PE2    (OFF_PE1 + 589824)
#define OFF_PE3    (OFF_PE2 + 589824)       // e3 packed: 8e*2kt*4nt*1024B = 65536

// ---------------- helpers ----------------
__device__ __forceinline__ unsigned short f2bf(float f) {
  unsigned int u = __builtin_bit_cast(unsigned int, f);
  u += 0x7FFFu + ((u >> 16) & 1u);           // round-to-nearest-even
  return (unsigned short)(u >> 16);
}
__device__ __forceinline__ float bf2f(unsigned short s) {
  unsigned int u = ((unsigned int)s) << 16;
  return __builtin_bit_cast(float, u);
}
__device__ __forceinline__ v16bf mk16(u16x8 lo, u16x8 hi) {
  u16x16 u = __builtin_shufflevector(lo, hi, 0,1,2,3,4,5,6,7,8,9,10,11,12,13,14,15);
  return __builtin_bit_cast(v16bf, u);
}
__device__ __forceinline__ v8f wmma_bf16(v16bf a, v16bf b, v8f c) {
  return __builtin_amdgcn_wmma_f32_16x16x32_bf16(false, a, false, b, (short)0, c,
                                                 false, false);
}
__device__ __forceinline__ void wait_asynccnt0() {
#if defined(__has_builtin)
#if __has_builtin(__builtin_amdgcn_s_wait_asynccnt)
  __builtin_amdgcn_s_wait_asynccnt(0);
  return;
#else
  asm volatile("s_wait_asynccnt 0x0" ::: "memory");
  return;
#endif
#else
  asm volatile("s_wait_asynccnt 0x0" ::: "memory");
#endif
}

// ---------------- kernel 1: zero pooled + gate scratch ----------------
__global__ void init_kernel(float* pooled) {
  int i = threadIdx.x;
  for (; i < 576; i += 256) pooled[i] = 0.f;   // 512 pooled + 64 expw (contiguous)
}

// ---------------- kernel 2: pack weights into WMMA B-fragment layout ----------
// B fragment (32x16, bf16): lane 0-15 -> col n = lane, halves h=0..15 -> K = kt*32+h
//                           lane 16-31 -> col n = lane-16, halves -> K = kt*32+16+h
// storage: frag-major, 512 halves per fragment, lane*16 + h within fragment.
__global__ void pack_weights(const float* __restrict__ c1w, const float* __restrict__ e1w,
                             const float* __restrict__ e2w, const float* __restrict__ e3w,
                             unsigned short* __restrict__ pc1, unsigned short* __restrict__ pe1,
                             unsigned short* __restrict__ pe2, unsigned short* __restrict__ pe3) {
  int gid = blockIdx.x * blockDim.x + threadIdx.x;
  if (gid >= 1360 * 512) return;
  int frag   = gid >> 9;
  int within = gid & 511;
  int lane   = within >> 4;
  int h      = within & 15;
  int kofs   = ((lane >= 16) ? 16 : 0) + h;
  int ncol   = lane & 15;
  float v; unsigned short* dst;
  if (frag < 144) {                                   // conv1: 18 kt x 8 nt
    int kt = frag >> 3, nt = frag & 7;
    int K = kt * 32 + kofs, n = nt * 16 + ncol;
    int cin = K & 63, tap = K >> 6;
    v = c1w[(n * 64 + cin) * 9 + tap];
    dst = pc1 + gid;
  } else if (frag < 144 + 576) {                      // e1: 8 e x 18 kt x 4 nt
    int f = frag - 144;
    int e = f / 72, r = f % 72, kt = r >> 2, nt = r & 3;
    int K = kt * 32 + kofs, n = nt * 16 + ncol;
    int cin = K & 63, tap = K >> 6;
    v = e1w[((e * 64 + n) * 64 + cin) * 9 + tap];
    dst = pe1 + (f * 512 + within);
  } else if (frag < 144 + 1152) {                     // e2
    int f = frag - (144 + 576);
    int e = f / 72, r = f % 72, kt = r >> 2, nt = r & 3;
    int K = kt * 32 + kofs, n = nt * 16 + ncol;
    int cin = K & 63, tap = K >> 6;
    v = e2w[((e * 64 + n) * 64 + cin) * 9 + tap];
    dst = pe2 + (f * 512 + within);
  } else {                                            // e3 (1x1): 8 e x 2 kt x 4 nt
    int f = frag - (144 + 1152);
    int e = f >> 3, r = f & 7, kt = r >> 2, nt = r & 3;
    int K = kt * 32 + kofs, n = nt * 16 + ncol;
    v = e3w[(e * 64 + n) * 64 + K];
    dst = pe3 + (f * 512 + within);
  }
  *dst = f2bf(v);
}

// ---------------- kernel 3: conv1 implicit GEMM + pooled sums ----------------
// grid: B * H * (W/64) = 2048 blocks, 128 thr (4 waves). Each wave: 16-pixel strip,
// N = 128 out channels as 8 WMMA accumulators, K = 576 (tap-major, cin-minor).
__global__ void __launch_bounds__(128)
conv1_kernel(const float* __restrict__ x, const float* __restrict__ c1b,
             const unsigned short* __restrict__ pc1,
             unsigned short* __restrict__ hws, float* __restrict__ pooled) {
  __shared__ __align__(16) unsigned short sx[3 * 66 * 64];   // [ky][col][cin] bf16
  int bi = blockIdx.x;
  int b  = bi >> 8;
  int t  = bi & 255;
  int y  = t >> 1;
  int x0 = (t & 1) * 64;
  int tid = threadIdx.x;
  // stage input tile (fp32 NCHW -> bf16 channel-last with halo, zero-padded).
  // consecutive lanes -> consecutive gx within a (row, cin) run: coalesced.
  for (int idx = tid; idx < 3 * 64 * 66; idx += 128) {
    int c   = idx % 66;
    int pos = idx / 66;
    int r   = pos >> 6, cin = pos & 63;
    int gy = y + r - 1, gx = x0 + c - 1;
    float v = 0.f;
    if ((unsigned)gy < 128u && (unsigned)gx < 128u)
      v = x[(((size_t)b * 64 + cin) * 128 + gy) * 128 + gx];
    sx[(r * 66 + c) * 64 + cin] = f2bf(v);
  }
  __syncthreads();

  int wave = tid >> 5, lane = tid & 31;
  int mbase = wave * 16, pix = lane & 15, hsel = (lane >> 4) & 1, mtop = hsel * 8;
  v8f acc[8];
  const v8f vz = {0.f,0.f,0.f,0.f,0.f,0.f,0.f,0.f};
  #pragma unroll
  for (int i = 0; i < 8; ++i) acc[i] = vz;

  for (int kt = 0; kt < KT3; ++kt) {
    int tap = kt >> 1, cinb = (kt & 1) * 32;
    int row = tap / 3, kx = tap % 3;
    int col = mbase + pix + kx;
    int lo  = cinb + hsel * 16;
    int sb  = (row * 66 + col) * 64 + lo;
    v16bf af = mk16(*(const u16x8*)&sx[sb], *(const u16x8*)&sx[sb + 8]);
    const unsigned short* bp = pc1 + (kt * 8) * 512 + lane * 16;
    #pragma unroll
    for (int nt = 0; nt < 8; ++nt) {
      v16bf bf = mk16(*(const u16x8*)bp, *(const u16x8*)(bp + 8));
      acc[nt] = wmma_bf16(af, bf, acc[nt]);
      bp += 512;
    }
  }
  // epilogue: bias, bf16 channel-last store, pooled atomic sums (x1 half only)
  #pragma unroll
  for (int nt = 0; nt < 8; ++nt) {
    int n = nt * 16 + pix;
    float bias = c1b[n];
    float colsum = 0.f;
    #pragma unroll
    for (int r = 0; r < 8; ++r) {
      int M = mtop + r;
      int gx = x0 + mbase + M;
      float v = acc[nt][r] + bias;
      hws[(((size_t)b * 128 + y) * 128 + gx) * 128 + n] = f2bf(v);
      colsum += v;
    }
    if (n < 64) atomicAdd(&pooled[b * 64 + n], colsum);
  }
}

// ---------------- kernel 4: gate (mean pool -> linear -> softmax -> top2) --------
__global__ void gate_kernel(const float* __restrict__ wet, const float* __restrict__ pooled,
                            float* __restrict__ expw) {
  __shared__ float lg[8][8];
  int tid = threadIdx.x;
  if (tid < 64) {
    int b = tid >> 3, e = tid & 7;
    float s = 0.f;
    for (int c = 0; c < 64; ++c) s += pooled[b * 64 + c] * wet[e * 64 + c];
    lg[b][e] = s * (1.f / (HH * WW));
  }
  __syncthreads();
  if (tid < 8) {
    int b = tid;
    float m = lg[b][0];
    for (int e = 1; e < 8; ++e) m = fmaxf(m, lg[b][e]);
    float p[8]; float sum = 0.f;
    for (int e = 0; e < 8; ++e) { p[e] = __expf(lg[b][e] - m); sum += p[e]; }
    float inv = 1.f / sum;
    for (int e = 0; e < 8; ++e) p[e] *= inv;
    int i1 = 0;
    for (int e = 1; e < 8; ++e) if (p[e] > p[i1]) i1 = e;
    int i2 = (i1 == 0) ? 1 : 0;
    for (int e = 0; e < 8; ++e) if (e != i1 && p[e] > p[i2]) i2 = e;
    for (int e = 0; e < 8; ++e) expw[b * 8 + e] = (e == i1 || e == i2) ? p[e] : 0.f;
  }
}

// ---------------- kernel 5: fused expert pipeline ----------------
// grid: 2048 strips. Stage x1/k1 tiles once (async global->LDS); per nonzero
// expert: a = wmma-conv3x3(x1,e1), b = wmma-conv3x3(k1,e2), m=(a+b1)(b+b2),
// y = wmma-1x1(m,e3)+b3, accO += w*y.  out = x1 + accO (LDS-transposed, NT stores).
__global__ void __launch_bounds__(128)
expert_kernel(const unsigned short* __restrict__ hws,
              const unsigned short* __restrict__ pe1, const unsigned short* __restrict__ pe2,
              const unsigned short* __restrict__ pe3,
              const float* __restrict__ e1b, const float* __restrict__ e2b,
              const float* __restrict__ e3b,
              const float* __restrict__ expw, float* __restrict__ out) {
  __shared__ __align__(16) unsigned short sx1[3 * 66 * 64];
  __shared__ __align__(16) unsigned short sk1[3 * 66 * 64];
  __shared__ __align__(16) unsigned short sm[4][16 * 64];   // per-wave m tile
  __shared__ __align__(16) float sout[64 * 68];             // transpose pad: stride 68
  __shared__ float sew[8];
  int bi = blockIdx.x;
  int b  = bi >> 8;
  int t  = bi & 255;
  int y  = t >> 1;
  int x0 = (t & 1) * 64;
  int tid = threadIdx.x;

  // ---- stage both halves of h (channel-last, 128ch contiguous per pixel) ----
  for (int task = tid; task < 198 * 16; task += 128) {
    int pos = task >> 4, chunk = task & 15;
    int r = pos / 66, c = pos % 66;
    int gy = y + r - 1, gx = x0 + c - 1;
    unsigned short* ldst = (chunk < 8)
        ? &sx1[(r * 66 + c) * 64 + chunk * 8]
        : &sk1[(r * 66 + c) * 64 + (chunk - 8) * 8];
    if ((unsigned)gy < 128u && (unsigned)gx < 128u) {
      const unsigned short* g =
          &hws[(((size_t)b * 128 + gy) * 128 + gx) * 128 + chunk * 8];
#if USE_ASYNC_STAGING
      // per-lane 16B async DMA into LDS (VDST = LDS byte addr = flat addr[31:0])
      unsigned ldsoff = (unsigned)(uintptr_t)ldst;
      asm volatile("global_load_async_to_lds_b128 %0, %1, off"
                   :: "v"(ldsoff), "v"((unsigned long long)(uintptr_t)g)
                   : "memory");
#else
      *(u16x8*)ldst = *(const u16x8*)g;
#endif
    } else {
      u16x8 z = {0,0,0,0,0,0,0,0};
      *(u16x8*)ldst = z;
    }
  }
#if USE_ASYNC_STAGING
  wait_asynccnt0();
#endif
  if (tid < 8) sew[tid] = expw[b * 8 + tid];
  __syncthreads();

  int wave = tid >> 5, lane = tid & 31;
  int mbase = wave * 16, pix = lane & 15, hsel = (lane >> 4) & 1, mtop = hsel * 8;
  const v8f vz = {0.f,0.f,0.f,0.f,0.f,0.f,0.f,0.f};
  v8f accO[4];
  #pragma unroll
  for (int i = 0; i < 4; ++i) accO[i] = vz;

  for (int e = 0; e < 8; ++e) {
    float w = sew[e];                 // uniform across block -> uniform branch
    if (w == 0.f) continue;           // only top-2 experts contribute (exact)
    v8f accA[4], accB[4];
    #pragma unroll
    for (int i = 0; i < 4; ++i) { accA[i] = vz; accB[i] = vz; }
    const unsigned short* pe1e = pe1 + (size_t)e * 72 * 512;
    const unsigned short* pe2e = pe2 + (size_t)e * 72 * 512;
    for (int kt = 0; kt < KT3; ++kt) {
      int tap = kt >> 1, cinb = (kt & 1) * 32;
      int row = tap / 3, kx = tap % 3;
      int col = mbase + pix + kx;
      int lo  = cinb + hsel * 16;
      int sb  = (row * 66 + col) * 64 + lo;
      v16bf a1 = mk16(*(const u16x8*)&sx1[sb], *(const u16x8*)&sx1[sb + 8]);
      v16bf a2 = mk16(*(const u16x8*)&sk1[sb], *(const u16x8*)&sk1[sb + 8]);
      const unsigned short* b1 = pe1e + kt * 4 * 512 + lane * 16;
      const unsigned short* b2 = pe2e + kt * 4 * 512 + lane * 16;
      #pragma unroll
      for (int nt = 0; nt < 4; ++nt) {
        v16bf bf1 = mk16(*(const u16x8*)b1, *(const u16x8*)(b1 + 8));
        v16bf bf2 = mk16(*(const u16x8*)b2, *(const u16x8*)(b2 + 8));
        accA[nt] = wmma_bf16(a1, bf1, accA[nt]);
        accB[nt] = wmma_bf16(a2, bf2, accB[nt]);
        b1 += 512; b2 += 512;
      }
    }
    // m = (a + e1_b) * (b + e2_b), spilled to LDS in A-fragment-friendly layout
    #pragma unroll
    for (int nt = 0; nt < 4; ++nt) {
      int n = nt * 16 + pix;
      float ab = e1b[e * 64 + n], bb = e2b[e * 64 + n];
      #pragma unroll
      for (int r = 0; r < 8; ++r) {
        float mv = (accA[nt][r] + ab) * (accB[nt][r] + bb);
        sm[wave][(mtop + r) * 64 + n] = f2bf(mv);
      }
    }
    __syncthreads();
    // y = m @ e3^T  (M=16, K=64, N=64)
    v8f accY[4];
    #pragma unroll
    for (int i = 0; i < 4; ++i) accY[i] = vz;
    #pragma unroll
    for (int kt2 = 0; kt2 < 2; ++kt2) {
      int lo = kt2 * 32 + hsel * 16;
      int sb = pix * 64 + lo;
      v16bf af = mk16(*(const u16x8*)&sm[wave][sb], *(const u16x8*)&sm[wave][sb + 8]);
      const unsigned short* b3 = pe3 + ((size_t)e * 8 + kt2 * 4) * 512 + lane * 16;
      #pragma unroll
      for (int nt = 0; nt < 4; ++nt) {
        v16bf bf = mk16(*(const u16x8*)b3, *(const u16x8*)(b3 + 8));
        accY[nt] = wmma_bf16(af, bf, accY[nt]);
        b3 += 512;
      }
    }
    #pragma unroll
    for (int nt = 0; nt < 4; ++nt) {
      int n = nt * 16 + pix;
      float b3v = e3b[e * 64 + n];
      #pragma unroll
      for (int r = 0; r < 8; ++r) accO[nt][r] += w * (accY[nt][r] + b3v);
    }
    __syncthreads();
  }

  // ---- residual, transpose through LDS, coalesced non-temporal NCHW stores ----
  #pragma unroll
  for (int nt = 0; nt < 4; ++nt) {
    int n = nt * 16 + pix;
    #pragma unroll
    for (int r = 0; r < 8; ++r) {
      int M = mtop + r;
      float x1v = bf2f(sx1[(66 + (mbase + M + 1)) * 64 + n]);
      sout[n * 68 + (mbase + M)] = x1v + accO[nt][r];
    }
  }
  __syncthreads();
  for (int tsk = tid; tsk < 64 * 16; tsk += 128) {
    int n = tsk >> 4, chunk = tsk & 15;
    f32x4 v4 = *(const f32x4*)&sout[n * 68 + chunk * 4];
    f32x4* dp = (f32x4*)&out[(((size_t)b * 64 + n) * 128 + y) * 128 + x0 + chunk * 4];
    __builtin_nontemporal_store(v4, dp);
  }
}

// ---------------- host ----------------
extern "C" void kernel_launch(void* const* d_in, const int* in_sizes, int n_in,
                              void* d_out, int out_size, void* d_ws, size_t ws_size,
                              hipStream_t stream) {
  const float* x   = (const float*)d_in[0];
  const float* c1w = (const float*)d_in[1];
  const float* c1b = (const float*)d_in[2];
  const float* wet = (const float*)d_in[3];
  const float* e1w = (const float*)d_in[4];
  const float* e1b = (const float*)d_in[5];
  const float* e2w = (const float*)d_in[6];
  const float* e2b = (const float*)d_in[7];
  const float* e3w = (const float*)d_in[8];
  const float* e3b = (const float*)d_in[9];
  float* out = (float*)d_out;
  char* ws = (char*)d_ws;
  unsigned short* hws = (unsigned short*)(ws + OFF_H);
  float* pooled = (float*)(ws + OFF_POOL);
  float* expw   = (float*)(ws + OFF_EXPW);
  unsigned short* pc1 = (unsigned short*)(ws + OFF_PC1);
  unsigned short* pe1 = (unsigned short*)(ws + OFF_PE1);
  unsigned short* pe2 = (unsigned short*)(ws + OFF_PE2);
  unsigned short* pe3 = (unsigned short*)(ws + OFF_PE3);

  hipLaunchKernelGGL(init_kernel,  dim3(1),    dim3(256), 0, stream, pooled);
  hipLaunchKernelGGL(pack_weights, dim3(2720), dim3(256), 0, stream,
                     c1w, e1w, e2w, e3w, pc1, pe1, pe2, pe3);
  hipLaunchKernelGGL(conv1_kernel, dim3(2048), dim3(128), 0, stream,
                     x, c1b, pc1, hws, pooled);
  hipLaunchKernelGGL(gate_kernel,  dim3(1),    dim3(64),  0, stream,
                     wet, pooled, expw);
  hipLaunchKernelGGL(expert_kernel, dim3(2048), dim3(128), 0, stream,
                     hws, pe1, pe2, pe3, e1b, e2b, e3b, expw, out);
}
